// QuasiLayer_26113401159818
// MI455X (gfx1250) — matrix operand order repaired and checked by reading, not compiled, not run on hardware
//
#include <hip/hip_runtime.h>
#include <hip/hip_bf16.h>

// Problem constants
#define BB    8
#define SS    2048
#define CIN   512
#define HH    512
#define N3H   1536      // 3*H
#define KDIM  1024      // CIN * K(=2)
#define MTOT  (BB*SS)   // 16384
#define SP1   (SS+1)    // padded time length (row 0 = zeros)

typedef unsigned short u16;
typedef unsigned int   u32;
typedef unsigned long long u64;

typedef __bf16 v16bf __attribute__((ext_vector_type(16)));
typedef float  v8f   __attribute__((ext_vector_type(8)));

// ---------- bf16 helpers (round-to-nearest-even) ----------
__device__ __forceinline__ u16 f2bf_rne(float f) {
    union { float f; u32 u; } v; v.f = f;
    u32 u = v.u;
    u32 r = u + 0x7FFFu + ((u >> 16) & 1u);
    return (u16)(r >> 16);
}
__device__ __forceinline__ float bf2f(u16 h) {
    union { u32 u; float f; } v; v.u = ((u32)h) << 16;
    return v.f;
}

union FragU { uint4 q[2]; v16bf v; };

// ---------- CDNA5 async memory->LDS staging (no VGPR data path) ----------
// GV form: per-lane 16B, LDS dest addr in VGPR, 64-bit global addr in VGPR pair.
__device__ __forceinline__ void async_to_lds_b128(u32 lds_byte_off, const void* gptr) {
    asm volatile("global_load_async_to_lds_b128 %0, %1, off"
                 :: "v"(lds_byte_off), "v"((u64)(size_t)gptr)
                 : "memory");
}
__device__ __forceinline__ void wait_async0() {
#if __has_builtin(__builtin_amdgcn_s_wait_asynccnt)
    __builtin_amdgcn_s_wait_asynccnt(0);
#else
    asm volatile("s_wait_asynccnt 0x0" ::: "memory");
#endif
}

// ---------------------------------------------------------------------------
// Kernel 1: conv_weight (3H, CIN, 2) fp32 -> n-major k-contiguous bf16 hi/lo:
// Wp[n*1024 + k], k = tap*512 + cin  (tap 0 multiplies x[t-1]).
// ---------------------------------------------------------------------------
__global__ void qrnn_prep_weights(const float* __restrict__ W,
                                  u16* __restrict__ Whi,
                                  u16* __restrict__ Wlo) {
    u32 idx = blockIdx.x * 256u + threadIdx.x;          // [0, 1536*1024)
    u32 n = idx >> 10;
    u32 k = idx & 1023u;
    u32 cin = k & 511u;
    u32 tap = k >> 9;
    float w = W[(size_t)n * 1024u + cin * 2u + tap];
    u16 hi = f2bf_rne(w);
    u16 lo = f2bf_rne(w - bf2f(hi));
    Whi[idx] = hi;
    Wlo[idx] = lo;
}

// ---------------------------------------------------------------------------
// Kernel 2: input (B,S,CIN) fp32 -> bf16 hi/lo with one zero timestep
// prepended per batch:  Xp[b][j][cin] = (j==0) ? 0 : x[b][j-1][cin].
// ---------------------------------------------------------------------------
__global__ void qrnn_prep_x(const float* __restrict__ x,
                            u16* __restrict__ Xhi,
                            u16* __restrict__ Xlo) {
    u32 idx = blockIdx.x * 256u + threadIdx.x;      // [0, 8*2049*512)
    u32 b   = idx / (SP1 * CIN);
    u32 rem = idx - b * (SP1 * CIN);
    u32 j   = rem >> 9;
    u32 cin = rem & 511u;
    float v = 0.0f;
    if (j != 0)
        v = x[((size_t)b * SS + (j - 1)) * CIN + cin];
    u16 hi = f2bf_rne(v);
    u16 lo = f2bf_rne(v - bf2f(hi));
    Xhi[idx] = hi;
    Xlo[idx] = lo;
}

// ---------------------------------------------------------------------------
// Kernel 3: gates GEMM (M=16384, N=1536, K=1024), bf16 hi/lo split-precision:
//   acc += Ahi*Bhi + Ahi*Blo + Alo*Bhi   (fp32 accumulate)
// Block 256 thr = 8 waves; block tile 128M x 128N; wave tile 32M x 64N.
// Staging: global_load_async_to_lds_b128 into double-buffered dynamic LDS
// (64 KB, plane map below); one s_wait_asynccnt + barrier per k-step.
//
// Dynamic LDS byte map (assumes dynamic LDS starts at offset 0; no static LDS):
//   A planes: stage s, plane p (0=hi,1=lo) at  s*16384 + p*8192     (32 KB)
//   B planes: 32768 + s*16384 + p*8192                              (32 KB)
// Each plane = 128 rows x 32 bf16 (row-major, 64 B rows).
// ---------------------------------------------------------------------------
__global__ __launch_bounds__(256)
void qrnn_gates_gemm(const u16* __restrict__ Xhi,
                     const u16* __restrict__ Xlo,
                     const u16* __restrict__ Whi,
                     const u16* __restrict__ Wlo,
                     const float* __restrict__ bias,
                     float* __restrict__ gates) {
    extern __shared__ u16 smem[];      // 32768 u16 = 64 KB

    const int tid  = threadIdx.x;
    const int lane = tid & 31;
    const int wave = tid >> 5;
    const int wm   = wave & 3;         // 4 M sub-blocks of 32 rows
    const int wn   = wave >> 2;        // 2 N halves of 64 cols
    const int half = lane >> 4;
    const int l15  = lane & 15;

    const int mBase = blockIdx.x * 128;    // 128 blocks
    const int nBase = blockIdx.y * 128;    // 12 blocks

    v8f acc[2][4];
#pragma unroll
    for (int s = 0; s < 2; ++s)
#pragma unroll
        for (int a = 0; a < 4; ++a) acc[s][a] = (v8f)0.0f;

    // Issue the 8 async 16B transfers this thread owns for k-step `ks`
    // into LDS stage `s`. 256 threads cover all 4 planes (32 KB/stage).
    auto issueStage = [&](int ks, int s) {
        const int kcol = ks * 32;
        const int tap  = (ks >= 16) ? 1 : 0;
        const int cin0 = kcol & 511;
        const u32 sbase = (u32)s * 16384u;
#pragma unroll
        for (int it = 0; it < 2; ++it) {
            int c   = tid + it * 256;      // 512 chunks of 16 B per plane
            int row = c >> 2;              // 0..127
            int c4  = c & 3;               // 0..3
            int m = mBase + row;
            int b = m >> 11;
            int t = m & (SS - 1);
            size_t aoff = ((size_t)b * SP1 + t + tap) * CIN + cin0 + c4 * 8;
            size_t boff = (size_t)(nBase + row) * KDIM + kcol + c4 * 8;
            u32 loff = (u32)(row * 64 + c4 * 16);     // bytes within plane
            async_to_lds_b128(sbase + loff,                 Xhi + aoff);
            async_to_lds_b128(sbase + 8192u + loff,         Xlo + aoff);
            async_to_lds_b128(32768u + sbase + loff,        Whi + boff);
            async_to_lds_b128(32768u + sbase + 8192u + loff, Wlo + boff);
        }
    };

    // ---- software pipeline ----
    issueStage(0, 0);
    wait_async0();
    __syncthreads();

    for (int ks = 0; ks < 32; ++ks) {
        const int cur = ks & 1;
        if (ks < 31) issueStage(ks + 1, cur ^ 1);   // background copy

        const u16* sAhi = smem + cur * 8192;            // elements
        const u16* sAlo = sAhi + 4096;
        const u16* sBhi = smem + 16384 + cur * 8192;
        const u16* sBlo = sBhi + 4096;

        // A fragments for both 16-row sub-tiles (ISA 16-bit A 16x32 layout)
        FragU ah[2], al[2];
#pragma unroll
        for (int ms = 0; ms < 2; ++ms) {
            const int arow = (wm * 32 + ms * 16 + l15) * 32;
            ah[ms].q[0] = *reinterpret_cast<const uint4*>(&sAhi[arow + 8 * half]);
            ah[ms].q[1] = *reinterpret_cast<const uint4*>(&sAhi[arow + 16 + 8 * half]);
            al[ms].q[0] = *reinterpret_cast<const uint4*>(&sAlo[arow + 8 * half]);
            al[ms].q[1] = *reinterpret_cast<const uint4*>(&sAlo[arow + 16 + 8 * half]);
        }

#pragma unroll
        for (int a = 0; a < 4; ++a) {
            const int brow = (wn * 64 + a * 16 + l15) * 32;
            FragU bh, bl;
            bh.q[0] = *reinterpret_cast<const uint4*>(&sBhi[brow + 8 * half]);
            bh.q[1] = *reinterpret_cast<const uint4*>(&sBhi[brow + 16 + 8 * half]);
            bl.q[0] = *reinterpret_cast<const uint4*>(&sBlo[brow + 8 * half]);
            bl.q[1] = *reinterpret_cast<const uint4*>(&sBlo[brow + 16 + 8 * half]);
#pragma unroll
            for (int ms = 0; ms < 2; ++ms) {
                acc[ms][a] = __builtin_amdgcn_wmma_f32_16x16x32_bf16(
                    false, ah[ms].v, false, bh.v, (short)0, acc[ms][a], false, false);
                acc[ms][a] = __builtin_amdgcn_wmma_f32_16x16x32_bf16(
                    false, ah[ms].v, false, bl.v, (short)0, acc[ms][a], false, false);
                acc[ms][a] = __builtin_amdgcn_wmma_f32_16x16x32_bf16(
                    false, al[ms].v, false, bh.v, (short)0, acc[ms][a], false, false);
            }
        }

        if (ks < 31) {
            wait_async0();       // this wave's copies into stage cur^1 done
            __syncthreads();     // everyone's copies visible
        }
    }

    // ---- epilogue: bias + activation, write z/f/o planes [g][B][S][H] ----
#pragma unroll
    for (int ms = 0; ms < 2; ++ms) {
#pragma unroll
        for (int a = 0; a < 4; ++a) {
#pragma unroll
            for (int r = 0; r < 8; ++r) {
                int mloc = wm * 32 + ms * 16 + r + 8 * half;
                int nloc = wn * 64 + a * 16 + l15;
                int m = mBase + mloc;
                int n = nBase + nloc;
                float val = acc[ms][a][r] + bias[n];
                int g  = n >> 9;            // 0:z(tanh) 1:f 2:o (sigmoid)
#if __has_builtin(__builtin_amdgcn_tanhf)
                if (g == 0) val = __builtin_amdgcn_tanhf(val);
                else        val = 0.5f + 0.5f * __builtin_amdgcn_tanhf(0.5f * val);
#else
                if (g == 0) val = tanhf(val);
                else        val = 1.0f / (1.0f + __expf(-val));
#endif
                int hh = n & 511;
                int b = m >> 11;
                int t = m & (SS - 1);
                gates[(((size_t)g * BB + b) * SS + t) * HH + hh] = val;
            }
        }
    }
}

// ---------------------------------------------------------------------------
// Kernel 4: sequential scan.  c = f*c + (1-f)*z ; h = o*c.
// One thread per (b, channel): 4096 threads, coalesced over channel.
// ---------------------------------------------------------------------------
__global__ __launch_bounds__(256)
void qrnn_scan(const float* __restrict__ gates, float* __restrict__ out) {
    const size_t plane = (size_t)BB * SS * HH;
    int tid = blockIdx.x * 256 + threadIdx.x;    // 0..4095
    int b  = tid >> 9;
    int hh = tid & 511;
    const float* Z = gates;
    const float* F = gates + plane;
    const float* O = gates + 2 * plane;
    size_t base = (size_t)b * SS * HH + hh;
    float c = 0.0f;
    float h = 0.0f;
#pragma unroll 4
    for (int t = 0; t < SS; ++t) {
        size_t idx = base + (size_t)t * HH;
        float f = F[idx];
        float z = Z[idx];
        float o = O[idx];
        c = f * c + (1.0f - f) * z;
        h = o * c;
        out[idx] = h;
    }
    out[plane + (size_t)b * HH + hh] = h;   // h_t
}

// ---------------------------------------------------------------------------
extern "C" void kernel_launch(void* const* d_in, const int* in_sizes, int n_in,
                              void* d_out, int out_size, void* d_ws, size_t ws_size,
                              hipStream_t stream) {
    const float* x    = (const float*)d_in[0];   // (B,S,CIN)
    const float* W    = (const float*)d_in[1];   // (3H,CIN,2)
    const float* bias = (const float*)d_in[2];   // (3H,)
    float* out = (float*)d_out;

    // ws layout: [gates 3*B*S*H f32][Whi][Wlo bf16][Xhi][Xlo bf16 padded]
    const size_t gates_elems = (size_t)3 * BB * SS * HH;   // 25,165,824 f32
    const size_t w_elems     = (size_t)N3H * KDIM;         // 1,572,864
    const size_t x_elems     = (size_t)BB * SP1 * CIN;     // 8,392,704
    char* ws = (char*)d_ws;
    float* gates = (float*)ws;
    u16* Whi = (u16*)(ws + gates_elems * sizeof(float));
    u16* Wlo = Whi + w_elems;
    u16* Xhi = Wlo + w_elems;
    u16* Xlo = Xhi + x_elems;

    // 1) weight split/relayout
    qrnn_prep_weights<<<(N3H * KDIM) / 256, 256, 0, stream>>>(W, Whi, Wlo);
    // 2) input split + causal zero-pad
    qrnn_prep_x<<<(BB * SP1 * CIN) / 256, 256, 0, stream>>>(x, Xhi, Xlo);
    // 3) gates GEMM: grid (M/128, N/128) = (128, 12), 64 KB dynamic LDS
    dim3 grid(MTOT / 128, N3H / 128);
    qrnn_gates_gemm<<<grid, 256, 65536, stream>>>(Xhi, Xlo, Whi, Wlo, bias, gates);
    // 4) recurrence scan: B*H = 4096 threads
    qrnn_scan<<<(BB * HH) / 256, 256, 0, stream>>>(gates, out);
}
